// Seq2SeqWithCNNEncoder_42253888258613
// MI455X (gfx1250) — matrix operand order repaired and checked by reading, not compiled, not run on hardware
//
#include <hip/hip_runtime.h>
#include <cstddef>

typedef __bf16 bf16;
typedef __attribute__((ext_vector_type(16))) __bf16 v16bf;
typedef __attribute__((ext_vector_type(8)))  __bf16 v8bf;
typedef __attribute__((ext_vector_type(8)))  float  v8f;
typedef __attribute__((ext_vector_type(4)))  unsigned int u32x4;

#define B_SZ   64
#define T_ENC  2048
#define T_DEC  128

__device__ __forceinline__ float sigm(float x) { return 1.0f / (1.0f + __expf(-x)); }
// overflow-safe fast tanh on v_exp_f32: x->+inf => 1, x->-inf => -1
__device__ __forceinline__ float tanh_fast(float x) {
  float e = __expf(2.0f * x);
  return 1.0f - 2.0f / (e + 1.0f);
}

// ---------------------------------------------------------------------------
// CDNA5 async global->LDS DMA (ASYNCcnt-tracked), per-lane 16B.
// VDST VGPR holds the LDS byte address (low 32 bits of generic shared ptr).
// ---------------------------------------------------------------------------
__device__ __forceinline__ void async_ld_b128(unsigned ldsOff, const void* gaddr) {
  asm volatile("global_load_async_to_lds_b128 %0, %1, off"
               :: "v"(ldsOff), "v"((unsigned long long)(size_t)gaddr)
               : "memory");
}
__device__ __forceinline__ void wait_async0() {
  asm volatile("s_wait_asynccnt 0x0" ::: "memory");
}

// ---------------------------------------------------------------------------
// Weight packing: W [N,K] f32 row-major -> B-fragment packed bf16 for B = W^T
// tile (nt,kt) at ((nt*Ktiles)+kt)*512 halfs; element (k_local,n_local):
//   lane = n_local + 16*((k_local>>3)&1), half = (k_local & 7) + 8*(k_local>>4)
// ---------------------------------------------------------------------------
__global__ void pack_b(const float* __restrict__ W, bf16* __restrict__ P, int N, int K) {
  int idx = blockIdx.x * 256 + threadIdx.x;
  if (idx >= N * K) return;
  int half = idx & 15;
  int lane = (idx >> 4) & 31;
  int tile = idx >> 9;
  int Kt = K >> 5;
  int nt = tile / Kt, kt = tile % Kt;
  int n = nt * 16 + (lane & 15);
  int k = kt * 32 + (lane >> 4) * 8 + (half & 7) + 16 * (half >> 3);
  P[idx] = (bf16)W[(long)n * K + k];
}

// conv2 weights [O=256, I=256, KK=3] -> packed B with K=768 (k = kk*256 + i), N=256
__global__ void pack_b_conv2(const float* __restrict__ W2, bf16* __restrict__ P) {
  int idx = blockIdx.x * 256 + threadIdx.x;
  const int N = 256, K = 768;
  if (idx >= N * K) return;
  int half = idx & 15;
  int lane = (idx >> 4) & 31;
  int tile = idx >> 9;
  int Kt = K >> 5;
  int nt = tile / Kt, kt = tile % Kt;
  int n = nt * 16 + (lane & 15);
  int k = kt * 32 + (lane >> 4) * 8 + (half & 7) + 16 * (half >> 3);
  int kk = k / 256, i = k % 256;
  P[idx] = (bf16)W2[(long)n * 768 + i * 3 + kk];
}

__global__ void vec_add(const float* __restrict__ a, const float* __restrict__ b,
                        float* __restrict__ o, int n) {
  int i = blockIdx.x * 256 + threadIdx.x;
  if (i < n) o[i] = a[i] + b[i];
}

// ---------------------------------------------------------------------------
// conv1 (1 -> 256 channels, k=3, SAME) + ReLU, output padded [B, T+2, 256] bf16
// ---------------------------------------------------------------------------
__global__ void conv1_kernel(const float* __restrict__ enc, const float* __restrict__ w,
                             const float* __restrict__ bias, bf16* __restrict__ y1p, int T) {
  int t = blockIdx.x;          // 0 .. T+1
  int b = blockIdx.y;
  int o = threadIdx.x;         // 0 .. 255
  long row = ((long)b * (T + 2) + t) * 256;
  if (t == 0 || t == T + 1) { y1p[row + o] = (bf16)0.0f; return; }
  int tt = t - 1;
  float x0 = (tt - 1 >= 0) ? enc[(long)b * T + tt - 1] : 0.0f;
  float x1 = enc[(long)b * T + tt];
  float x2 = (tt + 1 < T) ? enc[(long)b * T + tt + 1] : 0.0f;
  float v = bias[o] + w[o * 3 + 0] * x0 + w[o * 3 + 1] * x1 + w[o * 3 + 2] * x2;
  y1p[row + o] = (bf16)fmaxf(v, 0.0f);
}

// ---------------------------------------------------------------------------
// Generic WMMA GEMM: C = act(A @ Bpacked + bias), bf16 in/out, f32 accum.
// Block = 256 thr (8 waves). Workgroup tile 64(M) x 128(N), k-step 32.
// A staged global->LDS via async DMA, double-buffered (1 barrier / k-step);
// B fragments rotated one k-step ahead so loads overlap the WMMA pipe.
// permT>0: out_row = (r % permT)*permB + r/permT  (used for [B*T]->[T][B])
// ---------------------------------------------------------------------------
__global__ void __launch_bounds__(256) gemm_bf16_wmma(
    const bf16* __restrict__ A, int lda,
    const bf16* __restrict__ Bp,
    const float* __restrict__ bias,
    bf16* __restrict__ C,
    int N, int Kdim, int relu,
    int permT, int permB,
    long zAstride, int zOutRows) {
  __shared__ __align__(32) bf16 As[2][4 * 512];

  const int tid  = threadIdx.x;
  const int lane = tid & 31;
  const int w    = tid >> 5;        // 0..7
  const int mTile = w & 3;
  const int nGrp  = w >> 2;         // 0..1
  const long mBase = (long)blockIdx.y * 64;
  const int  Ktiles = Kdim >> 5;
  const bf16* Ab = A + (long)blockIdx.z * zAstride;

  v8f acc[4];
#pragma unroll
  for (int j = 0; j < 4; ++j)
#pragma unroll
    for (int i = 0; i < 8; ++i) acc[j][i] = 0.0f;

  // async staging plan: thread -> (row r, 8-half chunk c) -> A-fragment slot
  const int r   = tid >> 2;
  const int cch = tid & 3;
  const int dLane = (r & 15) + 16 * (cch & 1);
  const int dHalf = 8 * (cch >> 1);
  const int dInBuf = (r >> 4) * 512 + dLane * 16 + dHalf;
  const unsigned dst0 = (unsigned)(size_t)&As[0][dInBuf];
  const unsigned dst1 = (unsigned)(size_t)&As[1][dInBuf];
  const bf16* srcRow = Ab + (mBase + r) * (long)lda + cch * 8;

  const int ntBase = blockIdx.x * 8 + nGrp * 4;
  const bf16* bpL = Bp + (long)ntBase * Ktiles * 512 + lane * 16;
  const long bStride = (long)Ktiles * 512;

  // prologue: stage k-step 0, preload its B fragments
  async_ld_b128(dst0, srcRow);
  v16bf bc0 = *(const v16bf*)(bpL + 0 * bStride);
  v16bf bc1 = *(const v16bf*)(bpL + 1 * bStride);
  v16bf bc2 = *(const v16bf*)(bpL + 2 * bStride);
  v16bf bc3 = *(const v16bf*)(bpL + 3 * bStride);
  wait_async0();
  __syncthreads();

  for (int ks = 0; ks < Ktiles; ++ks) {
    const bool more = (ks + 1 < Ktiles);
    if (more)
      async_ld_b128(((ks + 1) & 1) ? dst1 : dst0, srcRow + (ks + 1) * 32);
    v16bf bn0, bn1, bn2, bn3;
    if (more) {
      const bf16* bp = bpL + (long)(ks + 1) * 512;
      bn0 = *(const v16bf*)(bp + 0 * bStride);
      bn1 = *(const v16bf*)(bp + 1 * bStride);
      bn2 = *(const v16bf*)(bp + 2 * bStride);
      bn3 = *(const v16bf*)(bp + 3 * bStride);
    }
    v16bf a = *(const v16bf*)(&As[ks & 1][mTile * 512 + lane * 16]);
    acc[0] = __builtin_amdgcn_wmma_f32_16x16x32_bf16(false, a, false, bc0, (short)0, acc[0], false, false);
    acc[1] = __builtin_amdgcn_wmma_f32_16x16x32_bf16(false, a, false, bc1, (short)0, acc[1], false, false);
    acc[2] = __builtin_amdgcn_wmma_f32_16x16x32_bf16(false, a, false, bc2, (short)0, acc[2], false, false);
    acc[3] = __builtin_amdgcn_wmma_f32_16x16x32_bf16(false, a, false, bc3, (short)0, acc[3], false, false);
    if (more) { bc0 = bn0; bc1 = bn1; bc2 = bn2; bc3 = bn3; }
    wait_async0();
    __syncthreads();
  }

  // epilogue: C/D layout -> lane = n + 16*(m/8); vgpr j -> m = 8*(lane>>4)+j
  const int nl   = lane & 15;
  const int mOff = (lane >> 4) * 8;
  const long rowBase = mBase + mTile * 16 + mOff + (long)blockIdx.z * zOutRows;
#pragma unroll
  for (int j = 0; j < 4; ++j) {
    int n = (ntBase + j) * 16 + nl;
    float bv = bias ? bias[n] : 0.0f;
#pragma unroll
    for (int i = 0; i < 8; ++i) {
      float v = acc[j][i] + bv;
      if (relu) v = fmaxf(v, 0.0f);
      long rr = rowBase + i;
      long orow = permT ? ((rr % permT) * (long)permB + rr / permT) : rr;
      C[orow * N + n] = (bf16)v;
    }
  }
}

// ---------------------------------------------------------------------------
// Persistent encoder LSTM: 1 workgroup, 1024 threads = 32 waves.
// Per step: gates = Gin[t] + h @ Whh^T  (WMMA, M=64 N=1024 K=256),
// wave w owns m-tile (w&3) and n-tile columns {2q,2q+1} for ALL four gates
// (q = w>>2), so the nonlinearity is wave-private; c lives in registers.
// h is kept in LDS in A-fragment layout between steps. Gin[t+1] prefetched.
// ---------------------------------------------------------------------------
__global__ void __launch_bounds__(1024) lstm_encoder(
    const bf16* __restrict__ Gin,    // [T][64][1024] bf16 (bias folded in)
    const bf16* __restrict__ WhhP,   // packed K=256,N=1024
    bf16* __restrict__ encOut,       // [64][T][256]
    bf16* __restrict__ hOut, float* __restrict__ cOut, int T) {
  __shared__ __align__(32) bf16 hfrag[8 * 4 * 512];   // [kstep][mtile][frag]

  const int tid  = threadIdx.x;
  const int lane = tid & 31;
  const int w    = tid >> 5;       // 0..31
  const int mTile = w & 3;
  const int qq    = w >> 2;        // 0..7
  const int nl    = lane & 15;
  const int mOff  = (lane >> 4) * 8;

  for (int i = tid; i < 8 * 4 * 512; i += 1024) hfrag[i] = (bf16)0.0f;

  float cReg[2][8];
#pragma unroll
  for (int p = 0; p < 2; ++p)
#pragma unroll
    for (int i = 0; i < 8; ++i) cReg[p][i] = 0.0f;

  for (int t = 0; t < T; ++t) {
    __syncthreads();                       // h writes (t-1) / init visible

    // warm L2/L0 for next step's 128KB gate block (1024 thr x 128B)
    if (t + 1 < T)
      __builtin_prefetch(Gin + ((long)(t + 1) * 64 * 1024) + tid * 64, 0, 3);

    v8f acc[2][4];
#pragma unroll
    for (int p = 0; p < 2; ++p)
#pragma unroll
      for (int g = 0; g < 4; ++g)
#pragma unroll
        for (int i = 0; i < 8; ++i) acc[p][g][i] = 0.0f;

    for (int ks = 0; ks < 8; ++ks) {
      v16bf a = *(const v16bf*)(&hfrag[(ks * 4 + mTile) * 512 + lane * 16]);
#pragma unroll
      for (int p = 0; p < 2; ++p) {
#pragma unroll
        for (int g = 0; g < 4; ++g) {
          int nt = (2 * qq + p) + g * 16;      // gate regions 16 n-tiles apart
          v16bf b = *(const v16bf*)(WhhP + ((long)nt * 8 + ks) * 512 + lane * 16);
          acc[p][g] = __builtin_amdgcn_wmma_f32_16x16x32_bf16(false, a, false, b, (short)0, acc[p][g], false, false);
        }
      }
    }
    __syncthreads();                       // all hfrag reads for step t done

    const bf16* gin = Gin + (long)t * 64 * 1024;
#pragma unroll
    for (int p = 0; p < 2; ++p) {
      int nh = (2 * qq + p) * 16 + nl;
#pragma unroll
      for (int i = 0; i < 8; ++i) {
        int m = mTile * 16 + mOff + i;
        const bf16* gm = gin + (long)m * 1024 + nh;
        float gi = acc[p][0][i] + (float)gm[0];
        float gf = acc[p][1][i] + (float)gm[256];
        float gg = acc[p][2][i] + (float)gm[512];
        float go = acc[p][3][i] + (float)gm[768];
        float c = sigm(gf) * cReg[p][i] + sigm(gi) * tanh_fast(gg);
        float h = sigm(go) * tanh_fast(c);
        cReg[p][i] = c;
        encOut[((long)m * T + t) * 256 + nh] = (bf16)h;
        // write h into A-fragment layout for next step
        int ks2 = nh >> 5, kl = nh & 31;
        int lt = (m & 15) + 16 * ((kl >> 3) & 1);
        int hf = (kl & 7) + 8 * (kl >> 4);
        hfrag[(ks2 * 4 + (m >> 4)) * 512 + lt * 16 + hf] = (bf16)h;
        if (t == T - 1) { hOut[m * 256 + nh] = (bf16)h; cOut[m * 256 + nh] = c; }
      }
    }
  }
}

// ---------------------------------------------------------------------------
// Decoder LSTMCell step (input size 1): g = prev*wih + cbias + h @ cWhh^T
// Single workgroup, WMMA tiling as encoder; h staged via async LDS DMA.
// ---------------------------------------------------------------------------
__global__ void __launch_bounds__(1024) dec_cell(
    const float* __restrict__ prev,   // [64] scalar inputs
    const float* __restrict__ wih,    // [1024] (cell_wih[:,0])
    const float* __restrict__ cbias,  // [1024] = cell_bih + cell_bhh
    const bf16* __restrict__ WhhP,    // packed K=256,N=1024
    bf16* __restrict__ hDec,          // [64,256] in/out bf16
    float* __restrict__ cDec) {       // [64,256] in/out f32
  __shared__ __align__(32) bf16 hfrag[8 * 4 * 512];

  const int tid  = threadIdx.x;
  const int lane = tid & 31;
  const int w    = tid >> 5;
  const int mTile = w & 3;
  const int qq    = w >> 2;
  const int nl    = lane & 15;
  const int mOff  = (lane >> 4) * 8;

  // stage h (row-major) into A-fragment layout: 2048 async 16B DMAs
#pragma unroll
  for (int u = 0; u < 2; ++u) {
    int ci = tid + u * 1024;
    int ks = ci >> 8, rem = ci & 255;
    int r = rem >> 2, c = rem & 3;
    int dLane = (r & 15) + 16 * (c & 1);
    int dHalf = 8 * (c >> 1);
    unsigned d = (unsigned)(size_t)&hfrag[(ks * 4 + (r >> 4)) * 512 + dLane * 16 + dHalf];
    async_ld_b128(d, hDec + r * 256 + ks * 32 + c * 8);
  }
  wait_async0();
  __syncthreads();

  v8f acc[2][4];
#pragma unroll
  for (int p = 0; p < 2; ++p)
#pragma unroll
    for (int g = 0; g < 4; ++g)
#pragma unroll
      for (int i = 0; i < 8; ++i) acc[p][g][i] = 0.0f;

  for (int ks = 0; ks < 8; ++ks) {
    v16bf a = *(const v16bf*)(&hfrag[(ks * 4 + mTile) * 512 + lane * 16]);
#pragma unroll
    for (int p = 0; p < 2; ++p) {
#pragma unroll
      for (int g = 0; g < 4; ++g) {
        int nt = (2 * qq + p) + g * 16;
        v16bf b = *(const v16bf*)(WhhP + ((long)nt * 8 + ks) * 512 + lane * 16);
        acc[p][g] = __builtin_amdgcn_wmma_f32_16x16x32_bf16(false, a, false, b, (short)0, acc[p][g], false, false);
      }
    }
  }

#pragma unroll
  for (int p = 0; p < 2; ++p) {
    int nh = (2 * qq + p) * 16 + nl;
#pragma unroll
    for (int i = 0; i < 8; ++i) {
      int m = mTile * 16 + mOff + i;
      float inp = prev[m];
      float gi = acc[p][0][i] + inp * wih[nh +   0] + cbias[nh +   0];
      float gf = acc[p][1][i] + inp * wih[nh + 256] + cbias[nh + 256];
      float gg = acc[p][2][i] + inp * wih[nh + 512] + cbias[nh + 512];
      float go = acc[p][3][i] + inp * wih[nh + 768] + cbias[nh + 768];
      float c = sigm(gf) * cDec[m * 256 + nh] + sigm(gi) * tanh_fast(gg);
      float h = sigm(go) * tanh_fast(c);
      cDec[m * 256 + nh] = c;
      hDec[m * 256 + nh] = (bf16)h;
    }
  }
}

// ---------------------------------------------------------------------------
// Attention: scores, softmax, context (batched GEMV, not WMMA-shaped)
// ---------------------------------------------------------------------------
__global__ void attn_scores(const bf16* __restrict__ qv, const bf16* __restrict__ Km,
                            float* __restrict__ s, int Tenc) {
  __shared__ bf16 qs[256];
  int b = blockIdx.y;
  int tid = threadIdx.x;
  qs[tid] = qv[b * 256 + tid];
  __syncthreads();
  int t = blockIdx.x * 256 + tid;
  const bf16* kr = Km + ((long)b * Tenc + t) * 256;
  float acc = 0.0f;
#pragma unroll 4
  for (int k = 0; k < 256; k += 8) {
    v8bf kk = *(const v8bf*)(kr + k);
    v8bf qq = *(const v8bf*)(&qs[k]);
#pragma unroll
    for (int j = 0; j < 8; ++j) acc += (float)kk[j] * (float)qq[j];
  }
  s[(long)b * Tenc + t] = acc * 0.0625f;   // 1/sqrt(256)
}

__global__ void softmax_rows(const float* __restrict__ s, float* __restrict__ p, int Tenc) {
  __shared__ float red[256];
  int b = blockIdx.x, tid = threadIdx.x;
  const float* sr = s + (long)b * Tenc;
  float* pr = p + (long)b * Tenc;
  float m = -1e30f;
  for (int i = tid; i < Tenc; i += 256) m = fmaxf(m, sr[i]);
  red[tid] = m; __syncthreads();
  for (int st = 128; st > 0; st >>= 1) {
    if (tid < st) red[tid] = fmaxf(red[tid], red[tid + st]);
    __syncthreads();
  }
  float mx = red[0]; __syncthreads();
  float sum = 0.0f;
  for (int i = tid; i < Tenc; i += 256) {
    float e = __expf(sr[i] - mx);
    pr[i] = e;
    sum += e;
  }
  red[tid] = sum; __syncthreads();
  for (int st = 128; st > 0; st >>= 1) {
    if (tid < st) red[tid] += red[tid + st];
    __syncthreads();
  }
  float inv = 1.0f / red[0];
  for (int i = tid; i < Tenc; i += 256) pr[i] *= inv;
}

__global__ void attn_ctx(const float* __restrict__ p, const bf16* __restrict__ Vm,
                         bf16* __restrict__ ctx, int Tenc) {
  __shared__ float pc[256];
  int b = blockIdx.x, e = threadIdx.x;
  float acc = 0.0f;
  for (int t0 = 0; t0 < Tenc; t0 += 256) {
    __syncthreads();
    pc[e] = p[(long)b * Tenc + t0 + e];
    __syncthreads();
    const bf16* vb = Vm + ((long)b * Tenc + t0) * 256 + e;
    for (int j = 0; j < 256; ++j) acc += pc[j] * (float)vb[j * 256];
  }
  ctx[b * 256 + e] = (bf16)acc;
}

// residual + LayerNorm + output projection (scalar) + feedback
__global__ void dec_out(const bf16* __restrict__ y0, const bf16* __restrict__ hDec,
                        const float* __restrict__ lnw, const float* __restrict__ lnb,
                        const float* __restrict__ outw, const float* __restrict__ outb,
                        float* __restrict__ dOut, float* __restrict__ prev,
                        int t, int Tdec) {
  int r = threadIdx.x;    // 0..63
  const bf16* yr = y0 + r * 256;
  const bf16* hr = hDec + r * 256;
  float mu = 0.0f;
  for (int e = 0; e < 256; ++e) mu += (float)yr[e] + (float)hr[e];
  mu *= (1.0f / 256.0f);
  float var = 0.0f;
  for (int e = 0; e < 256; ++e) {
    float d = (float)yr[e] + (float)hr[e] - mu;
    var += d * d;
  }
  var *= (1.0f / 256.0f);
  float inv = rsqrtf(var + 1e-5f);
  float o = outb[0];
  for (int e = 0; e < 256; ++e) {
    float yn = ((float)yr[e] + (float)hr[e] - mu) * inv * lnw[e] + lnb[e];
    o += yn * outw[e];
  }
  dOut[(long)r * Tdec + t] = o;
  prev[r] = o;
}

__global__ void init_prev(const float* __restrict__ dec, float* __restrict__ prev, int Tdec) {
  int b = threadIdx.x;
  prev[b] = dec[(long)b * Tdec];   // decoder_inputs[b, 0, 0]
}

// ---------------------------------------------------------------------------
extern "C" void kernel_launch(void* const* d_in, const int* in_sizes, int n_in,
                              void* d_out, int out_size, void* d_ws, size_t ws_size,
                              hipStream_t stream) {
  const float* enc      = (const float*)d_in[0];   // [64,2048,1]
  const float* dec      = (const float*)d_in[1];   // [64,128,1]
  const float* conv1_w  = (const float*)d_in[2];
  const float* conv1_b  = (const float*)d_in[3];
  const float* conv2_w  = (const float*)d_in[4];
  const float* conv2_b  = (const float*)d_in[5];
  const float* lstm_wih = (const float*)d_in[6];
  const float* lstm_whh = (const float*)d_in[7];
  const float* lstm_bih = (const float*)d_in[8];
  const float* lstm_bhh = (const float*)d_in[9];
  const float* cell_wih = (const float*)d_in[10];
  const float* cell_whh = (const float*)d_in[11];
  const float* cell_bih = (const float*)d_in[12];
  const float* cell_bhh = (const float*)d_in[13];
  const float* wq = (const float*)d_in[14]; const float* bq = (const float*)d_in[15];
  const float* wk = (const float*)d_in[16]; const float* bk = (const float*)d_in[17];
  const float* wv = (const float*)d_in[18]; const float* bv = (const float*)d_in[19];
  const float* wo = (const float*)d_in[20]; const float* bo = (const float*)d_in[21];
  const float* ln_w = (const float*)d_in[22]; const float* ln_b = (const float*)d_in[23];
  const float* out_w = (const float*)d_in[24]; const float* out_b = (const float*)d_in[25];
  float* outp = (float*)d_out;

  // ---- workspace bump allocation (256B aligned) ----
  char* ws = (char*)d_ws;
  size_t off = 0;
  auto alloc = [&](size_t bytes) -> void* {
    void* p = ws + off;
    off = (off + bytes + 255) & ~(size_t)255;
    return p;
  };
  bf16*  y1p    = (bf16*)alloc((size_t)B_SZ * (T_ENC + 2) * 256 * 2);
  bf16*  y2     = (bf16*)alloc((size_t)B_SZ * T_ENC * 256 * 2);
  bf16*  Gin    = (bf16*)alloc((size_t)T_ENC * B_SZ * 1024 * 2);
  bf16*  encOut = (bf16*)alloc((size_t)B_SZ * T_ENC * 256 * 2);
  bf16*  Km     = (bf16*)alloc((size_t)B_SZ * T_ENC * 256 * 2);
  bf16*  Vm     = (bf16*)alloc((size_t)B_SZ * T_ENC * 256 * 2);
  bf16*  w2p    = (bf16*)alloc(768 * 256 * 2);
  bf16*  wihp   = (bf16*)alloc(256 * 1024 * 2);
  bf16*  whhp   = (bf16*)alloc(256 * 1024 * 2);
  bf16*  cwhhp  = (bf16*)alloc(256 * 1024 * 2);
  bf16*  wqp    = (bf16*)alloc(256 * 256 * 2);
  bf16*  wkp    = (bf16*)alloc(256 * 256 * 2);
  bf16*  wvp    = (bf16*)alloc(256 * 256 * 2);
  bf16*  wop    = (bf16*)alloc(256 * 256 * 2);
  float* lbias  = (float*)alloc(1024 * 4);
  float* cbias  = (float*)alloc(1024 * 4);
  bf16*  hDec   = (bf16*)alloc(64 * 256 * 2);
  float* cDec   = (float*)alloc(64 * 256 * 4);
  bf16*  qbuf   = (bf16*)alloc(64 * 256 * 2);
  float* scores = (float*)alloc((size_t)64 * T_ENC * 4);
  float* probs  = (float*)alloc((size_t)64 * T_ENC * 4);
  bf16*  ctxb   = (bf16*)alloc(64 * 256 * 2);
  bf16*  y0     = (bf16*)alloc(64 * 256 * 2);
  float* prev   = (float*)alloc(64 * 4);
  (void)ws_size; (void)in_sizes; (void)n_in; (void)out_size;

  // ---- pack weights (deterministic each call) ----
  pack_b<<<(1024 * 256 + 255) / 256, 256, 0, stream>>>(lstm_wih, wihp, 1024, 256);
  pack_b<<<(1024 * 256 + 255) / 256, 256, 0, stream>>>(lstm_whh, whhp, 1024, 256);
  pack_b<<<(1024 * 256 + 255) / 256, 256, 0, stream>>>(cell_whh, cwhhp, 1024, 256);
  pack_b<<<(256 * 256 + 255) / 256, 256, 0, stream>>>(wq, wqp, 256, 256);
  pack_b<<<(256 * 256 + 255) / 256, 256, 0, stream>>>(wk, wkp, 256, 256);
  pack_b<<<(256 * 256 + 255) / 256, 256, 0, stream>>>(wv, wvp, 256, 256);
  pack_b<<<(256 * 256 + 255) / 256, 256, 0, stream>>>(wo, wop, 256, 256);
  pack_b_conv2<<<(768 * 256 + 255) / 256, 256, 0, stream>>>(conv2_w, w2p);
  vec_add<<<4, 256, 0, stream>>>(lstm_bih, lstm_bhh, lbias, 1024);
  vec_add<<<4, 256, 0, stream>>>(cell_bih, cell_bhh, cbias, 1024);

  // ---- encoder ----
  conv1_kernel<<<dim3(T_ENC + 2, B_SZ), 256, 0, stream>>>(enc, conv1_w, conv1_b, y1p, T_ENC);

  // conv2 as GEMM: per batch z, M=2048 rows (im2col window = 3 contiguous rows)
  gemm_bf16_wmma<<<dim3(256 / 128, T_ENC / 64, B_SZ), 256, 0, stream>>>(
      y1p, 256, w2p, conv2_b, y2, 256, 768, /*relu*/1, 0, 0,
      (long)(T_ENC + 2) * 256, T_ENC);

  // hoisted LSTM input gates: [B*T,256]@[256,1024], output [T][B][1024]
  gemm_bf16_wmma<<<dim3(1024 / 128, (B_SZ * T_ENC) / 64, 1), 256, 0, stream>>>(
      y2, 256, wihp, lbias, Gin, 1024, 256, 0, /*permT*/T_ENC, /*permB*/B_SZ, 0, 0);

  // sequential recurrence (persistent single-WGP kernel)
  lstm_encoder<<<1, 1024, 0, stream>>>(Gin, whhp, encOut, hDec, cDec, T_ENC);

  // K/V projections (hoisted out of decode loop)
  gemm_bf16_wmma<<<dim3(2, (B_SZ * T_ENC) / 64, 1), 256, 0, stream>>>(
      encOut, 256, wkp, bk, Km, 256, 256, 0, 0, 0, 0, 0);
  gemm_bf16_wmma<<<dim3(2, (B_SZ * T_ENC) / 64, 1), 256, 0, stream>>>(
      encOut, 256, wvp, bv, Vm, 256, 256, 0, 0, 0, 0, 0);

  // ---- decoder: 128 sequential steps ----
  init_prev<<<1, 64, 0, stream>>>(dec, prev, T_DEC);
  for (int t = 0; t < T_DEC; ++t) {
    dec_cell<<<1, 1024, 0, stream>>>(prev, cell_wih, cbias, cwhhp, hDec, cDec);
    gemm_bf16_wmma<<<dim3(2, 1, 1), 256, 0, stream>>>(
        hDec, 256, wqp, bq, qbuf, 256, 256, 0, 0, 0, 0, 0);
    attn_scores<<<dim3(T_ENC / 256, B_SZ), 256, 0, stream>>>(qbuf, Km, scores, T_ENC);
    softmax_rows<<<B_SZ, 256, 0, stream>>>(scores, probs, T_ENC);
    attn_ctx<<<B_SZ, 256, 0, stream>>>(probs, Vm, ctxb, T_ENC);
    gemm_bf16_wmma<<<dim3(2, 1, 1), 256, 0, stream>>>(
        ctxb, 256, wop, bo, y0, 256, 256, 0, 0, 0, 0, 0);
    dec_out<<<1, 64, 0, stream>>>(y0, hDec, ln_w, ln_b, out_w, out_b, outp, prev, t, T_DEC);
  }
}
